// HRNet_W48_Prob_Contrast_Proto_88691074663212
// MI455X (gfx1250) — matrix operand order repaired and checked by reading, not compiled, not run on hardware
//
#include <hip/hip_runtime.h>
#include <hip/hip_bf16.h>
#include <math.h>

// ---- problem constants --------------------------------------------------
#define K_CLS    19
#define M_PROTO  10
#define C_DIM    720
#define CP       736     // C padded to multiple of 32 (WMMA K-step)
#define CPL      744     // LDS row stride (bank-conflict padding)
#define P_TOT    190     // K_CLS * M_PROTO
#define PP       192     // padded proto rows (multiple of 16)
#define N_TILES  12      // PP / 16
#define WAVE_TILES 6     // tiles per wave (2 column-halves of 6)
#define INV_EPS  20.0f   // 1 / 0.05
#define ROWS_PER_BLOCK 128
#define GEMM_THREADS 512                // 16 waves: 8 row-slabs x 2 col-halves
#define GEMM_LDS_BYTES (PP * CPL * 2)   // 285,696 B  (<320KB WGP LDS)

typedef __bf16 bf16_t;
typedef __attribute__((ext_vector_type(8)))  __bf16 v8bf;
typedef __attribute__((ext_vector_type(16))) __bf16 v16bf;
typedef __attribute__((ext_vector_type(8)))  float  v8f;

union ABFrag { v16bf v; v8bf h[2]; };

// ---- block reduction (blockDim.x == 256) --------------------------------
__device__ __forceinline__ float block_reduce_sum(float v, float* sh) {
  int tid = threadIdx.x;
  sh[tid] = v; __syncthreads();
  for (int o = 128; o > 0; o >>= 1) {
    if (tid < o) sh[tid] += sh[tid + o];
    __syncthreads();
  }
  float r = sh[0];
  __syncthreads();
  return r;
}

// ---- kernel 1: zero the small reduction buffers -------------------------
__global__ void zero_small_kernel(float* __restrict__ colSums, float* __restrict__ countsF) {
  int i = threadIdx.x;
  for (int j = i; j < 3 * P_TOT; j += 256) colSums[j] = 0.0f;
  if (i < K_CLS) countsF[i] = 0.0f;
}

// ---- kernel 2: zero-fill the dense q output (mostly zeros) --------------
__global__ void zero_q_kernel(float4* __restrict__ q4, size_t n4) {
  size_t i = (size_t)blockIdx.x * blockDim.x + threadIdx.x;
  size_t stride = (size_t)gridDim.x * blockDim.x;
  for (; i < n4; i += stride) q4[i] = make_float4(0.f, 0.f, 0.f, 0.f);
}

// ---- kernel 3: L2-normalize prototypes -> bf16, padded ------------------
__global__ void proto_norm_kernel(const float* __restrict__ protos, bf16_t* __restrict__ protoBf) {
  __shared__ float sh[256];
  int p = blockIdx.x;           // 0..PP-1
  int tid = threadIdx.x;
  float ss = 0.0f;
  if (p < P_TOT) {
    for (int c = tid; c < C_DIM; c += 256) {
      float v = protos[(size_t)p * C_DIM + c];
      ss += v * v;
    }
  }
  float tot = block_reduce_sum(ss, sh);
  float inv = 1.0f / fmaxf(sqrtf(tot), 1e-12f);
  for (int c = tid; c < CP; c += 256) {
    float v = (p < P_TOT && c < C_DIM) ? protos[(size_t)p * C_DIM + c] * inv : 0.0f;
    protoBf[(size_t)p * CP + c] = (bf16_t)v;
  }
}

// ---- kernel 4: per-row LayerNorm + L2-normalize x -> bf16, padded -------
__global__ void x_norm_kernel(const float* __restrict__ x, const float* __restrict__ g,
                              const float* __restrict__ b, bf16_t* __restrict__ xnBf) {
  __shared__ float sh[256];
  int row = blockIdx.x;
  int tid = threadIdx.x;
  const float* xr = x + (size_t)row * C_DIM;
  float s = 0.0f, s2 = 0.0f;
  float xv[3]; int cnt = 0;
  for (int c = tid; c < C_DIM; c += 256) {
    float v = xr[c];
    xv[cnt++] = v; s += v; s2 += v * v;
  }
  float sum   = block_reduce_sum(s, sh);
  float sumsq = block_reduce_sum(s2, sh);
  float mu   = sum * (1.0f / C_DIM);
  float var  = sumsq * (1.0f / C_DIM) - mu * mu;
  float rstd = rsqrtf(var + 1e-5f);
  float yv[3]; float t = 0.0f; cnt = 0;
  for (int c = tid; c < C_DIM; c += 256) {
    float y = (xv[cnt] - mu) * rstd * g[c] + b[c];
    yv[cnt++] = y; t += y * y;
  }
  float l2  = block_reduce_sum(t, sh);
  float inv = 1.0f / fmaxf(sqrtf(l2), 1e-12f);
  bf16_t* outr = xnBf + (size_t)row * CP;
  cnt = 0;
  for (int c = tid; c < CP; c += 256) {
    float v = (c < C_DIM) ? (yv[cnt++] * inv) : 0.0f;
    outr[c] = (bf16_t)v;
  }
}

// ---- kernel 5: WMMA GEMM  sim[n, m*19+k] = xn . protoN^T ----------------
// CDNA5/WGP strategy: the full 192x736 bf16 B panel (282KB) is staged once
// into the WGP's 320KB LDS and shared by all 16 waves across all 23 K-steps.
// 16 waves = 8 row-slabs x 2 column-halves; each wave keeps only 6
// accumulators (48 VGPRs) + 6 B fragments, so the allocator can pin the
// accumulators (no WMMA->VALU copy/nop hazards) and keep all 6 B fragments
// in flight (staggered s_wait_dscnt instead of full waits).
__global__ void __launch_bounds__(GEMM_THREADS)
gemm_wmma_kernel(const bf16_t* __restrict__ xnBf, const bf16_t* __restrict__ protoBf,
                 float* __restrict__ simOut) {
  extern __shared__ char smemRaw[];
  bf16_t* smB = (bf16_t*)smemRaw;   // [PP][CPL]

  // ---- cooperative stage of the B panel into LDS (b128 granularity) ----
  {
    const int CHUNKS_PER_ROW = CP / 8;                 // 92 x v8bf
    for (int i = threadIdx.x; i < PP * CHUNKS_PER_ROW; i += GEMM_THREADS) {
      int r = i / CHUNKS_PER_ROW;
      int c = i - r * CHUNKS_PER_ROW;
      *reinterpret_cast<v8bf*>(smB + r * CPL + c * 8) =
          *reinterpret_cast<const v8bf*>(protoBf + (size_t)r * CP + c * 8);
    }
  }
  __syncthreads();

  int lane  = threadIdx.x & 31;
  int wave  = threadIdx.x >> 5;    // 0..15
  int mwave = wave & 7;            // row slab
  int cside = wave >> 3;           // column half: tiles [6*cside, 6*cside+6)
  int l15   = lane & 15;
  int half  = lane >> 4;
  size_t rowA = (size_t)blockIdx.x * ROWS_PER_BLOCK + mwave * 16 + l15;
  const bf16_t* aRow  = xnBf + rowA * CP;
  const bf16_t* bBase = smB + (cside * WAVE_TILES * 16 + l15) * CPL + half * 8;
  int selA = half * 8;

  v8f acc[WAVE_TILES];
  #pragma unroll
  for (int nt = 0; nt < WAVE_TILES; ++nt)
    #pragma unroll
    for (int r = 0; r < 8; ++r) acc[nt][r] = 0.0f;

  for (int kb = 0; kb < CP / 32; ++kb) {
    int kbase = kb * 32;
    // A fragment: two contiguous 16B chunks (K = half*8 and 16+half*8)
    ABFrag A;
    A.h[0] = *reinterpret_cast<const v8bf*>(aRow + kbase + selA);
    A.h[1] = *reinterpret_cast<const v8bf*>(aRow + kbase + 16 + selA);
    // All 6 B fragments into distinct registers (ds_load_b128 x2 each)
    ABFrag B[WAVE_TILES];
    #pragma unroll
    for (int nt = 0; nt < WAVE_TILES; ++nt) {
      const bf16_t* bRow = bBase + nt * 16 * CPL + kbase;
      B[nt].h[0] = *reinterpret_cast<const v8bf*>(bRow);
      B[nt].h[1] = *reinterpret_cast<const v8bf*>(bRow + 16);
    }
    #pragma unroll
    for (int nt = 0; nt < WAVE_TILES; ++nt) {
      acc[nt] = __builtin_amdgcn_wmma_f32_16x16x32_bf16(
          false, A.v, false, B[nt].v, (short)0, acc[nt], false, false);
    }
  }

  int rowBase = blockIdx.x * ROWS_PER_BLOCK + mwave * 16;
  #pragma unroll
  for (int nt = 0; nt < WAVE_TILES; ++nt) {
    int p = (cside * WAVE_TILES + nt) * 16 + l15;   // proto index = k*10 + m
    if (p < P_TOT) {
      int k = p / M_PROTO, m = p - k * M_PROTO;
      int col = m * K_CLS + k;                       // sim_mat is [n, m, k]
      #pragma unroll
      for (int r = 0; r < 8; ++r) {
        int row = rowBase + r + half * 8;
        simOut[(size_t)row * P_TOT + col] = acc[nt][r];
      }
    }
  }
}

// ---- kernel 6: per-pixel max/argmax, u = exp(sim/eps), first col-sums ---
__global__ void pixel_init_kernel(const float* __restrict__ sim, const int* __restrict__ gt,
                                  float* __restrict__ predOut, float* __restrict__ u,
                                  float* __restrict__ colSum0, float* __restrict__ countsF,
                                  int N) {
  __shared__ float sCol[P_TOT];
  __shared__ float sCnt[K_CLS];
  int tid = threadIdx.x;
  for (int i = tid; i < P_TOT; i += 256) sCol[i] = 0.0f;
  if (tid < K_CLS) sCnt[tid] = 0.0f;
  __syncthreads();
  int n = blockIdx.x * 256 + tid;
  if (n < N) {
    const float* row = sim + (size_t)n * P_TOT;
    int g = gt[n];
    float best = -1e30f; int karg = 0;
    for (int k = 0; k < K_CLS; ++k) {
      float mk = -1e30f;
      #pragma unroll
      for (int m = 0; m < M_PROTO; ++m) mk = fmaxf(mk, row[m * K_CLS + k]);
      if (mk > best) { best = mk; karg = k; }
    }
    predOut[n] = (float)karg;
    #pragma unroll
    for (int m = 0; m < M_PROTO; ++m) {
      float uv = __expf(row[m * K_CLS + g] * INV_EPS);
      u[(size_t)n * M_PROTO + m] = uv;
      atomicAdd(&sCol[g * M_PROTO + m], uv);
    }
    atomicAdd(&sCnt[g], 1.0f);
  }
  __syncthreads();
  for (int i = tid; i < P_TOT; i += 256)
    if (sCol[i] != 0.0f) atomicAdd(&colSum0[i], sCol[i]);
  if (tid < K_CLS && sCnt[tid] != 0.0f) atomicAdd(&countsF[tid], sCnt[tid]);
}

// ---- kernel 7: one Sinkhorn iteration (relaunched 3x for global sync) ---
__global__ void sk_iter_kernel(float* __restrict__ u, const int* __restrict__ gt,
                               const float* __restrict__ colSumIn, float* __restrict__ colSumOut,
                               const float* __restrict__ countsF, float* __restrict__ qOut,
                               int N, int last) {
  __shared__ float sCol[P_TOT];
  int tid = threadIdx.x;
  if (!last) for (int i = tid; i < P_TOT; i += 256) sCol[i] = 0.0f;
  __syncthreads();
  int n = blockIdx.x * 256 + tid;
  if (n < N) {
    int g = gt[n];
    float r[M_PROTO]; float rs = 0.0f;
    #pragma unroll
    for (int m = 0; m < M_PROTO; ++m) {
      float v = u[(size_t)n * M_PROTO + m]
              / fmaxf(colSumIn[g * M_PROTO + m], 1e-12f) * (1.0f / M_PROTO);
      r[m] = v; rs += v;
    }
    rs = fmaxf(rs, 1e-12f);
    if (last) {
      // final q = (u / colsum / M / rowsum / n) * n  ->  the /n * n cancels
      float* q = qOut + (size_t)g * N * M_PROTO + (size_t)n * M_PROTO;
      #pragma unroll
      for (int m = 0; m < M_PROTO; ++m) q[m] = r[m] / rs;
    } else {
      float nk = fmaxf(countsF[g], 1.0f);
      float inv = 1.0f / (rs * nk);
      #pragma unroll
      for (int m = 0; m < M_PROTO; ++m) {
        float v = r[m] * inv;
        u[(size_t)n * M_PROTO + m] = v;
        atomicAdd(&sCol[g * M_PROTO + m], v);
      }
    }
  }
  __syncthreads();
  if (!last)
    for (int i = tid; i < P_TOT; i += 256)
      if (sCol[i] != 0.0f) atomicAdd(&colSumOut[i], sCol[i]);
}

// ---- host launcher ------------------------------------------------------
extern "C" void kernel_launch(void* const* d_in, const int* in_sizes, int n_in,
                              void* d_out, int out_size, void* d_ws, size_t ws_size,
                              hipStream_t stream) {
  const float* x      = (const float*)d_in[0];
  const int*   gt     = (const int*)  d_in[1];
  const float* gamma  = (const float*)d_in[2];
  const float* beta   = (const float*)d_in[3];
  const float* protos = (const float*)d_in[4];
  const int N = in_sizes[0] / C_DIM;   // 65536

  // workspace layout (offsets 16B aligned except countsF)
  char* ws = (char*)d_ws;
  bf16_t* protoBf = (bf16_t*)ws;                                   // PP*CP bf16
  bf16_t* xnBf    = (bf16_t*)(ws + (size_t)PP * CP * 2);           // N*CP bf16
  size_t off = (size_t)PP * CP * 2 + (size_t)N * CP * 2;
  float* u        = (float*)(ws + off); off += (size_t)N * M_PROTO * 4;
  float* colSums  = (float*)(ws + off); off += (size_t)3 * P_TOT * 4;
  float* countsF  = (float*)(ws + off);

  // output layout: sim_mat [N,10,19] | q [19,N,10] | pred_seg [N]
  float* simOut  = (float*)d_out;
  float* qOut    = (float*)d_out + (size_t)N * P_TOT;
  float* predOut = (float*)d_out + 2 * (size_t)N * P_TOT;

  zero_small_kernel<<<1, 256, 0, stream>>>(colSums, countsF);
  size_t q4 = ((size_t)K_CLS * N * M_PROTO) / 4;
  zero_q_kernel<<<2048, 256, 0, stream>>>((float4*)qOut, q4);

  proto_norm_kernel<<<PP, 256, 0, stream>>>(protos, protoBf);
  x_norm_kernel<<<N, 256, 0, stream>>>(x, gamma, beta, xnBf);
  gemm_wmma_kernel<<<N / ROWS_PER_BLOCK, GEMM_THREADS, GEMM_LDS_BYTES, stream>>>(
      xnBf, protoBf, simOut);
  pixel_init_kernel<<<(N + 255) / 256, 256, 0, stream>>>(simOut, gt, predOut, u,
                                                         colSums, countsF, N);
  sk_iter_kernel<<<(N + 255) / 256, 256, 0, stream>>>(u, gt, colSums, colSums + P_TOT,
                                                      countsF, nullptr, N, 0);
  sk_iter_kernel<<<(N + 255) / 256, 256, 0, stream>>>(u, gt, colSums + P_TOT,
                                                      colSums + 2 * P_TOT, countsF,
                                                      nullptr, N, 0);
  sk_iter_kernel<<<(N + 255) / 256, 256, 0, stream>>>(u, gt, colSums + 2 * P_TOT, nullptr,
                                                      countsF, qOut, N, 1);
}
